// PSTConv_52913997087087
// MI455X (gfx1250) — matrix operand order repaired and checked by reading, not compiled, not run on hardware
//
#include <hip/hip_runtime.h>

typedef __attribute__((ext_vector_type(16))) _Float16 v16h;
typedef __attribute__((ext_vector_type(8)))  float    v8f;
typedef __attribute__((ext_vector_type(2)))  float    v2f;

#define BATCH 4
#define LFRM  8
#define NPTS  2048
#define CIN   32
#define MID   64
#define OUTC  128
#define KNB   16
#define MANCH 1024
#define CH    192           // TK * MID
#define R2    0.25f
#define BN_EPS 1e-5f

#if __has_builtin(__builtin_amdgcn_wmma_f32_16x16x4_f32)
#define HAVE_WMMA_F32X4 1
#else
#define HAVE_WMMA_F32X4 0
#endif

// Raw v_max_num_f32 without IEEE canonicalization preamble: inputs here are
// WMMA accumulator outputs / affine results, never signaling NaNs.
__device__ __forceinline__ float max_raw(float a, float b) {
  float d;
  asm("v_max_num_f32 %0, %1, %2" : "=v"(d) : "v"(a), "v"(b));
  return d;
}

// ---------------------------------------------------------------------------
// 1) Furthest point sampling: one 256-thread workgroup per (b, frame).
//    Point cloud in LDS, per-thread distances in registers, LDS tree argmax.
// ---------------------------------------------------------------------------
__global__ void PST_fps_kernel(const float* __restrict__ xyzs,
                               float* __restrict__ outXyz) {
  const int bf = blockIdx.x;            // b*8 + f
  const int b = bf >> 3, f = bf & 7;
  const float* fx = xyzs + ((size_t)(b * LFRM + f)) * NPTS * 3;
  __shared__ float sx[NPTS], sy[NPTS], sz[NPTS];
  __shared__ float rv[256];
  __shared__ int   ri[256];
  const int t = threadIdx.x;

  for (int n = t; n < NPTS; n += 256) {
    sx[n] = fx[n * 3 + 0];
    sy[n] = fx[n * 3 + 1];
    sz[n] = fx[n * 3 + 2];
  }
  __syncthreads();

  float* outp = outXyz + ((size_t)(b * LFRM + f)) * MANCH * 3;
  if (t == 0) { outp[0] = sx[0]; outp[1] = sy[0]; outp[2] = sz[0]; }

  float dloc[NPTS / 256];
#pragma unroll
  for (int j = 0; j < NPTS / 256; ++j) dloc[j] = 1e10f;

  int cur = 0;
  const int base = t * (NPTS / 256);
  for (int m = 1; m < MANCH; ++m) {
    const float cx = sx[cur], cy = sy[cur], cz = sz[cur];
    float bestv = -1.0f; int besti = base;
#pragma unroll
    for (int j = 0; j < NPTS / 256; ++j) {
      const int n = base + j;
      const float dx = sx[n] - cx, dy = sy[n] - cy, dz = sz[n] - cz;
      float d = dx * dx + dy * dy + dz * dz;
      d = fminf(dloc[j], d);
      dloc[j] = d;
      if (d > bestv) { bestv = d; besti = n; }   // strict > : lowest index wins
    }
    rv[t] = bestv; ri[t] = besti;
    __syncthreads();
    for (int s = 128; s > 0; s >>= 1) {
      if (t < s) {
        const float ov = rv[t + s]; const int oi = ri[t + s];
        if (ov > rv[t] || (ov == rv[t] && oi < ri[t])) { rv[t] = ov; ri[t] = oi; }
      }
      __syncthreads();
    }
    cur = ri[0];
    __syncthreads();
    if (t == 0) {
      outp[m * 3 + 0] = sx[cur];
      outp[m * 3 + 1] = sy[cur];
      outp[m * 3 + 2] = sz[cur];
    }
  }
}

// ---------------------------------------------------------------------------
// 2) Ball query: one wave per (b,tt,i,m) query; wave32 ballot-based
//    first-K-in-index-order selection matching pointnet2 semantics.
// ---------------------------------------------------------------------------
__global__ void PST_ballq_kernel(const float* __restrict__ xyzs,
                                 const float* __restrict__ anchors,
                                 int* __restrict__ ballIdx) {
  const int warp = blockIdx.x * (blockDim.x >> 5) + (threadIdx.x >> 5);
  const int lane = threadIdx.x & 31;
  const int m = warp & (MANCH - 1);
  int q = warp >> 10;                 // /MANCH
  const int i = q % 3; q /= 3;
  const int tt = q & 7; const int b = q >> 3;

  const float* a = anchors + (((size_t)(b * LFRM + tt)) * MANCH + m) * 3;
  const float ax = a[0], ay = a[1], az = a[2];
  int* out = ballIdx + (size_t)warp * KNB;
  const int f = tt + i - 1;

  if (f < 0 || f >= LFRM) {
    // zero-padded frame: all neighbors at origin
    const bool in = (ax * ax + ay * ay + az * az) < R2;
    if (lane < KNB) out[lane] = in ? lane : 0;
    return;
  }

  const float* nx = xyzs + ((size_t)(b * LFRM + f)) * NPTS * 3;
  int cnt = 0, firstIdx = 0;
  for (int n0 = 0; n0 < NPTS && cnt < KNB; n0 += 32) {
    const int n = n0 + lane;
    const float dx = nx[n * 3 + 0] - ax;
    const float dy = nx[n * 3 + 1] - ay;
    const float dz = nx[n * 3 + 2] - az;
    const bool in = (dx * dx + dy * dy + dz * dz) < R2;
    const unsigned mm = (unsigned)__ballot(in);
    if (cnt == 0 && mm) firstIdx = n0 + __ffs((int)mm) - 1;
    if (in) {
      const int slot = cnt + __popc(mm & ((1u << lane) - 1u));
      if (slot < KNB) out[slot] = n;
    }
    cnt += __popc(mm);
  }
  if (cnt < KNB && lane < KNB && lane >= cnt) out[lane] = firstIdx;
}

// ---------------------------------------------------------------------------
// 3) Grouped conv + max over K neighbors. Zero LDS, zero barriers:
//    each lane's B-fragment halves are exactly the values it gathers itself
//    (lane L: column = L%16, K-range = (L<16 ? 0..15 : 16..31)), so features
//    go straight from global memory into WMMA operand registers.
//    Wf term: V_WMMA_F32_16X16X32_F16 (K=32 == CIN, one op per row tile).
//    Wd@disp term: V_WMMA_F32_16X16X4_F32 (K=3 padded to 4, full f32).
//    Running max uses raw v_max_num_f32 (no canonicalization chains).
// ---------------------------------------------------------------------------
__global__ void PST_conv_kernel(const float* __restrict__ xyzs,
                                const float* __restrict__ feats,
                                const float* __restrict__ Wd,
                                const float* __restrict__ Wf,
                                const float* __restrict__ anchors,
                                const int* __restrict__ ballIdx,
                                float* __restrict__ sf) {
  const int bid = blockIdx.x;
  const int tg = bid & 7;
  int q = bid >> 3;
  const int i = q % 3; q /= 3;
  const int tt = q & 7; const int b = q >> 3;

  const int w = threadIdx.x >> 5;       // wave in block (0..7)
  const int lane = threadIdx.x & 31;
  const int hi = lane >> 4;             // 0: lanes 0-15, 1: lanes 16-31
  const int Mr = lane & 15;
  const int col = lane & 15;
  const int m0 = (tg * 8 + w) * 16;
  const int f = tt + i - 1;
  const bool valid = (f >= 0 && f < LFRM);

  // --- A fragments: Wf row-tiles (16x32 f16) ---
  v16h Af[4];
#pragma unroll
  for (int rt = 0; rt < 4; ++rt) {
#pragma unroll
    for (int h = 0; h < 16; ++h) {
      const int Kf = h + hi * 16;    // contiguous K layout (matches B)
      Af[rt][h] = (_Float16)Wf[(rt * 16 + Mr) * CIN + Kf];
    }
  }

#if HAVE_WMMA_F32X4
  // Wd row-tiles as 16x4 f32 A fragments (K=3 padded): 2 VGPRs per tile.
  // 32-bit A layout: lanes 0-15 hold K={0,1}, lanes 16-31 hold K={2,3}.
  v2f Ad[4];
#pragma unroll
  for (int rt = 0; rt < 4; ++rt) {
    const int c = rt * 16 + Mr;
    Ad[rt][0] = Wd[c * 3 + (hi ? 2 : 0)];
    Ad[rt][1] = hi ? 0.0f : Wd[c * 3 + 1];
  }
#else
  v16h Ad[4];
#pragma unroll
  for (int rt = 0; rt < 4; ++rt) {
#pragma unroll
    for (int h = 0; h < 16; ++h) {
      const int Kf = (h < 8 ? h : h + 8) + hi * 8;
      Ad[rt][h] = (Kf < 3) ? (_Float16)Wd[(rt * 16 + Mr) * 3 + Kf]
                           : (_Float16)0.0f;
    }
  }
#endif

  // anchor coords for this lane's column (lanes L and L+16 share a column)
  const float* ap = anchors + (((size_t)(b * LFRM + tt)) * MANCH + m0 + col) * 3;
  const float ax = ap[0], ay = ap[1], az = ap[2];

  const int* idxp =
      ballIdx + (((size_t)((b * LFRM + tt) * 3 + i)) * MANCH + m0 + col) * KNB;
  __builtin_prefetch(idxp, 0, 0);                       // global_prefetch_b8
  const float* fbase = feats + ((size_t)(b * LFRM + f)) * CIN * NPTS;
  const float* xbase = xyzs + ((size_t)(b * LFRM + f)) * NPTS * 3;
  const int chb = hi * 16;

  v8f vmax[4];
#pragma unroll
  for (int rt = 0; rt < 4; ++rt)
#pragma unroll
    for (int r = 0; r < 8; ++r) vmax[rt][r] = -3.0e38f;
  const v8f czero = {};

  for (int k = 0; k < KNB; ++k) {
    const int n = idxp[k];

    // B fragment (feat): this lane's 16 halves == the 16 channels it gathers
    v16h bfeat;
#pragma unroll
    for (int j = 0; j < 16; ++j) {
      const float v = valid ? fbase[(size_t)(chb + j) * NPTS + n] : 0.0f;
      bfeat[j] = (_Float16)v;
    }

    // displacement for this lane's column
    float gx = 0.f, gy = 0.f, gz = 0.f;
    if (valid) {
      const float* p = xbase + (size_t)n * 3;
      gx = p[0]; gy = p[1]; gz = p[2];
    }
    const float dx = gx - ax, dy = gy - ay, dz = gz - az;

#if HAVE_WMMA_F32X4
    // B 4x16 f32 fragment: lanes 0-15 hold K={0,1}, lanes 16-31 K={2,3}
    v2f bd;
    bd[0] = hi ? dz : dx;
    bd[1] = hi ? 0.0f : dy;
#else
    v16h bd;
#pragma unroll
    for (int h = 0; h < 16; ++h) bd[h] = (_Float16)0.0f;
    if (!hi) { bd[0] = (_Float16)dx; bd[1] = (_Float16)dy; bd[2] = (_Float16)dz; }
#endif

#pragma unroll
    for (int rt = 0; rt < 4; ++rt) {
#if HAVE_WMMA_F32X4
      v8f acc = __builtin_amdgcn_wmma_f32_16x16x4_f32(
          false, Ad[rt], false, bd, (short)0, czero, false, false);
#else
      v8f acc = __builtin_amdgcn_wmma_f32_16x16x32_f16(
          false, Ad[rt], false, bd, (short)0, czero, false, false);
#endif
      acc = __builtin_amdgcn_wmma_f32_16x16x32_f16(
          false, Af[rt], false, bfeat, (short)0, acc, false, false);
#pragma unroll
      for (int r = 0; r < 8; ++r) vmax[rt][r] = max_raw(vmax[rt][r], acc[r]);
    }
  }

  // write max-pooled result into sf[b,tt, i*64 + c, m]
#pragma unroll
  for (int rt = 0; rt < 4; ++rt) {
#pragma unroll
    for (int r = 0; r < 8; ++r) {
      const int ch = i * MID + rt * 16 + r + hi * 8;
      sf[(((size_t)(b * LFRM + tt)) * CH + ch) * MANCH + m0 + col] = vmax[rt][r];
    }
  }
}

// ---------------------------------------------------------------------------
// 4) BatchNorm statistics over (B, M) per (tt, channel); emit fused affine.
// ---------------------------------------------------------------------------
__global__ void PST_bnstats_kernel(const float* __restrict__ sf,
                                   const float* __restrict__ gamma,
                                   const float* __restrict__ beta,
                                   float* __restrict__ aArr,
                                   float* __restrict__ bArr) {
  const int tt = blockIdx.x / CH;
  const int ch = blockIdx.x % CH;
  __shared__ float ssum[256], ssq[256];
  const int t = threadIdx.x;
  float s = 0.f, sq = 0.f;
  for (int v = t; v < BATCH * MANCH; v += 256) {
    const int b = v >> 10;
    const int m = v & (MANCH - 1);
    const float x = sf[(((size_t)(b * LFRM + tt)) * CH + ch) * MANCH + m];
    s += x; sq += x * x;
  }
  ssum[t] = s; ssq[t] = sq;
  __syncthreads();
  for (int st = 128; st > 0; st >>= 1) {
    if (t < st) { ssum[t] += ssum[t + st]; ssq[t] += ssq[t + st]; }
    __syncthreads();
  }
  if (t == 0) {
    const float inv = 1.0f / (BATCH * MANCH);
    const float mean = ssum[0] * inv;
    const float var = ssq[0] * inv - mean * mean;
    const float rstd = rsqrtf(var + BN_EPS);
    const float g = gamma[ch] * rstd;
    aArr[tt * CH + ch] = g;
    bArr[tt * CH + ch] = beta[ch] - mean * g;
  }
}

// ---------------------------------------------------------------------------
// 5) Final GEMM: out[128,1024] = Wt[128,192] @ relu(bn(sf)), per (b,tt).
//    Block = 16 output columns; 8 waves cover the 8 row tiles and share the
//    LDS-staged normalized activations (B fragments identical across waves).
//    K=192 -> 6 chained 16x16x32 f16 WMMAs.
// ---------------------------------------------------------------------------
__global__ void PST_final_kernel(const float* __restrict__ sf,
                                 const float* __restrict__ Wt,
                                 const float* __restrict__ aArr,
                                 const float* __restrict__ bArr,
                                 float* __restrict__ outF) {
  const int bid = blockIdx.x;
  const int ntile = bid & 63;
  int q = bid >> 6;
  const int tt = q & 7; const int b = q >> 3;
  const int m0 = ntile * 16;
  const int t = threadIdx.x;

  __shared__ _Float16 sfT[16][CH];   // [col][K], K-contiguous per column

  for (int e = t; e < 16 * CH; e += 256) {
    const int ch = e >> 4;
    const int col = e & 15;
    const float x = sf[(((size_t)(b * LFRM + tt)) * CH + ch) * MANCH + m0 + col];
    const float y = max_raw(x * aArr[tt * CH + ch] + bArr[tt * CH + ch], 0.0f);
    sfT[col][ch] = (_Float16)y;
  }
  __syncthreads();

  const int w = t >> 5;
  const int lane = t & 31;
  const int hi = lane >> 4;
  const int Mr = lane & 15;
  const int col = lane & 15;

  v8f acc = {};
#pragma unroll
  for (int kk = 0; kk < 6; ++kk) {
    v16h a, bv;
#pragma unroll
    for (int h = 0; h < 16; ++h) {
      const int Kf = kk * 32 + hi * 16 + h;
      a[h] = (_Float16)Wt[(w * 16 + Mr) * CH + Kf];
    }
    const int kb = kk * 32 + hi * 16;
#pragma unroll
    for (int h = 0; h < 16; ++h) bv[h] = sfT[col][kb + h];
    acc = __builtin_amdgcn_wmma_f32_16x16x32_f16(
        false, a, false, bv, (short)0, acc, false, false);
  }

#pragma unroll
  for (int r = 0; r < 8; ++r) {
    const int oc = w * 16 + r + hi * 8;
    outF[(((size_t)(b * LFRM + tt)) * OUTC + oc) * MANCH + m0 + col] = acc[r];
  }
}

// ---------------------------------------------------------------------------
extern "C" void kernel_launch(void* const* d_in, const int* in_sizes, int n_in,
                              void* d_out, int out_size, void* d_ws, size_t ws_size,
                              hipStream_t stream) {
  (void)in_sizes; (void)n_in; (void)out_size; (void)ws_size;
  const float* xyzs  = (const float*)d_in[0];
  const float* feats = (const float*)d_in[1];
  const float* Wd    = (const float*)d_in[2];
  const float* Wf    = (const float*)d_in[3];
  const float* Wt    = (const float*)d_in[4];
  const float* gamma = (const float*)d_in[5];
  const float* beta  = (const float*)d_in[6];

  float* outXyz = (float*)d_out;                                  // [4,8,1024,3]
  float* outF   = outXyz + (size_t)BATCH * LFRM * MANCH * 3;      // [4,8,128,1024]

  char* ws = (char*)d_ws;
  int*   ballIdx = (int*)ws;                                      // 6 MiB
  float* sf      = (float*)(ws + (size_t)6 * 1024 * 1024);        // 24 MiB
  float* aArr    = (float*)(ws + (size_t)30 * 1024 * 1024);       // 6 KiB
  float* bArr    = (float*)(ws + (size_t)30 * 1024 * 1024 + 8192);

  // 1) FPS: 32 serial chains, one workgroup each
  PST_fps_kernel<<<BATCH * LFRM, 256, 0, stream>>>(xyzs, outXyz);

  // 2) Ball query: 98304 queries, 8 waves per 256-thread block
  PST_ballq_kernel<<<(BATCH * LFRM * 3 * MANCH) / 8, 256, 0, stream>>>(
      xyzs, outXyz, ballIdx);

  // 3) Grouped conv + max (WMMA): 768 blocks x 8 waves x 16-anchor tiles
  PST_conv_kernel<<<BATCH * LFRM * 3 * 8, 256, 0, stream>>>(
      xyzs, feats, Wd, Wf, outXyz, ballIdx, sf);

  // 4) BN statistics per (tt, channel)
  PST_bnstats_kernel<<<LFRM * CH, 256, 0, stream>>>(sf, gamma, beta, aArr, bArr);

  // 5) Final 128x192x1024 GEMM (WMMA) fused with BN affine + ReLU
  PST_final_kernel<<<BATCH * LFRM * 64, 256, 0, stream>>>(sf, Wt, aArr, bArr, outF);
}
